// GenesisClassifier_5068061409909
// MI455X (gfx1250) — compile-verified, matrix operand order checked by reading, not run on hardware
//
#include <hip/hip_runtime.h>
#include <hip/hip_bf16.h>

#define GT   2048   // sequence length (fixed by reference)
#define KPAD 20     // K dim padded 18 -> 20 (5 chunks of 4 for f32 WMMA)

typedef float v2f  __attribute__((ext_vector_type(2)));
typedef float v8f  __attribute__((ext_vector_type(8)));
typedef _Float16 v16h __attribute__((ext_vector_type(16)));

__device__ __forceinline__ float fast_tanh(float v) {
#if __has_builtin(__builtin_amdgcn_tanhf)
    return __builtin_amdgcn_tanhf(v);   // CDNA5 v_tanh_f32 (trans unit)
#else
    float vc = fminf(fmaxf(v, -9.0f), 9.0f);
    float e  = __builtin_amdgcn_exp2f(vc * 2.885390081777927f); // 2*log2(e)
    return (e - 1.0f) * __builtin_amdgcn_rcpf(e + 1.0f);
#endif
}

// Build zero-padded readout weights Wrp[16][KPAD] (N padded 2->16, K 18->20).
__global__ void __launch_bounds__(320) genesis_pad_wr(
    const float* __restrict__ Wr, float* __restrict__ Wrp)
{
    int i = threadIdx.x;            // 0..319 == n*KPAD + k
    if (i >= 16 * KPAD) return;
    int n = i / KPAD, k = i % KPAD;
    Wrp[i] = (n < 2 && k < 18) ? Wr[n * 18 + k] : 0.0f;
}

// One thread per (batch, hemisphere): 8192 threads = 256 wave32s.
// State (9 floats) + per-hemi params (27 floats) live in registers.
__global__ void __launch_bounds__(64) genesis_scan(
    const float* __restrict__ x,  const float* __restrict__ Wp,
    const float* __restrict__ bp, const float* __restrict__ w,
    const float* __restrict__ u,  const float* __restrict__ b,
    const float* __restrict__ Wr, const float* __restrict__ br,
    float* __restrict__ feat, float* __restrict__ out,
    int Bn, int fused)
{
    int tid  = blockIdx.x * blockDim.x + threadIdx.x;
    int bidx = tid >> 1;
    int h    = tid & 1;
    if (bidx >= Bn) return;

    const float Wp00 = Wp[0];
    const float bp0  = bp[0];

    float wv[9], uv[9], bv[9];
#pragma unroll
    for (int i = 0; i < 9; ++i) {
        wv[i] = w[h * 9 + i];
        uv[i] = u[h * 9 + i];
        bv[i] = b[h * 9 + i];
    }

    float z[9];
#pragma unroll
    for (int i = 0; i < 9; ++i) z[i] = 0.0f;

    const float* xr = x + (size_t)bidx * GT;   // 8KB-aligned row

    for (int t0 = 0; t0 < GT; t0 += 4) {
        float4 xv = *(const float4*)(xr + t0);
        __builtin_prefetch(xr + ((t0 + 8) & (GT - 1)), 0, 0);  // global_prefetch_b8
        float xs[4] = {xv.x, xv.y, xv.z, xv.w};
#pragma unroll
        for (int q = 0; q < 4; ++q) {
            float s = fmaf(xs[q], Wp00, bp0);
            // D module
            float zd0 = fast_tanh(fmaf(wv[0], s, fmaf(uv[0], z[0], bv[0])));
            float zd1 = fast_tanh(fmaf(wv[1], s, fmaf(uv[1], z[1], bv[1])));
            float zd2 = fast_tanh(fmaf(wv[2], s, fmaf(uv[2], z[2], bv[2])));
            float ti  = (zd0 + zd1 + zd2) * (1.0f / 3.0f);
            // T module
            float zt0 = fast_tanh(fmaf(wv[3], ti, fmaf(uv[3], z[3], bv[3])));
            float zt1 = fast_tanh(fmaf(wv[4], ti, fmaf(uv[4], z[4], bv[4])));
            float zt2 = fast_tanh(fmaf(wv[5], ti, fmaf(uv[5], z[5], bv[5])));
            float qi  = (zt0 + zt1 + zt2) * (1.0f / 3.0f);
            // Q module
            float zq0 = fast_tanh(fmaf(wv[6], qi, fmaf(uv[6], z[6], bv[6])));
            float zq1 = fast_tanh(fmaf(wv[7], qi, fmaf(uv[7], z[7], bv[7])));
            float zq2 = fast_tanh(fmaf(wv[8], qi, fmaf(uv[8], z[8], bv[8])));
            z[0] = zd0; z[1] = zd1; z[2] = zd2;
            z[3] = zt0; z[4] = zt1; z[5] = zt2;
            z[6] = zq0; z[7] = zq1; z[8] = zq2;
        }
    }

    if (!fused) {
        // padded row: [KPAD] floats; h==1 also zero-fills the K pad tail
        float* f = feat + (size_t)bidx * KPAD + h * 9;
#pragma unroll
        for (int i = 0; i < 9; ++i) f[i] = z[i];
        if (h) { f[9] = 0.0f; f[10] = 0.0f; }   // k = 18, 19
    } else {
        // fallback readout (only if ws too small): cross-hemi sum via wave32 shuffle
#pragma unroll
        for (int n = 0; n < 2; ++n) {
            float p = 0.0f;
#pragma unroll
            for (int i = 0; i < 9; ++i) p = fmaf(z[i], Wr[n * 18 + h * 9 + i], p);
            p += __shfl_xor(p, 1);
            if (h == 0) out[(size_t)bidx * 2 + n] = p + br[n];
        }
    }
}

// Readout: out[Bn,2] = feat[Bn,18] @ Wr^T[18,2] + br via f32 WMMA.
// One wave per 16 batch rows. Inputs pre-padded in memory -> branch-free
// loop: 10x global_load_b64 + 5x v_wmma_f32_16x16x4_f32, EXEC untouched.
__global__ void __launch_bounds__(256) genesis_readout_wmma(
    const float* __restrict__ featp, const float* __restrict__ Wrp,
    const float* __restrict__ br, float* __restrict__ out, int Bn)
{
    int gtid    = blockIdx.x * blockDim.x + threadIdx.x;
    int wave    = gtid >> 5;
    int lane    = threadIdx.x & 31;
    int rowBase = wave * 16;
    if (rowBase >= Bn) return;   // uniform per wave

    int m    = lane & 15;   // A: row M; B: col N; D: col N
    int half = lane >> 4;

    const float* frow = featp + (size_t)(rowBase + m) * KPAD;  // 80B stride, 8B aligned
    const float* brow = Wrp + m * KPAD;
    v8f acc = {};

#if __has_builtin(__builtin_amdgcn_wmma_f32_16x16x4_f32)
    // A 16x4 f32 per lane: {K=half*2, K=half*2+1}; B 4x16 mirrored.
#pragma unroll
    for (int kc = 0; kc < 5; ++kc) {
        int k0 = kc * 4 + half * 2;
        v2f a  = *(const v2f*)(frow + k0);
        v2f bm = *(const v2f*)(brow + k0);
        acc = __builtin_amdgcn_wmma_f32_16x16x4_f32(
            false, a, false, bm, (short)0, acc, false, false);
    }
#else
    // Fallback: single f16 16x16x32 WMMA (K=32 >= 18), codegen-confirmed builtin.
    v16h a, bm;
#pragma unroll
    for (int r = 0; r < 8; ++r) {
        int kb = (r < 4) ? (r * 2 + half * 8) : (16 + (r - 4) * 2 + half * 8);
        a[2 * r]      = (kb     < 18) ? (_Float16)frow[kb]     : (_Float16)0.0f;
        a[2 * r + 1]  = (kb + 1 < 18) ? (_Float16)frow[kb + 1] : (_Float16)0.0f;
        bm[2 * r]     = (kb     < 18) ? (_Float16)brow[kb]     : (_Float16)0.0f;
        bm[2 * r + 1] = (kb + 1 < 18) ? (_Float16)brow[kb + 1] : (_Float16)0.0f;
    }
    acc = __builtin_amdgcn_wmma_f32_16x16x32_f16(
        false, a, false, bm, (short)0, acc, false, false);
#endif

    if (m < 2) {
        float bias = br[m];
#pragma unroll
        for (int v = 0; v < 8; ++v) {
            int M = v + half * 8;   // D layout: VGPR v -> row M = v (+8 upper halfwave)
            out[(size_t)(rowBase + M) * 2 + m] = acc[v] + bias;
        }
    }
}

extern "C" void kernel_launch(void* const* d_in, const int* in_sizes, int n_in,
                              void* d_out, int out_size, void* d_ws, size_t ws_size,
                              hipStream_t stream) {
    const float* x  = (const float*)d_in[0];
    const float* Wp = (const float*)d_in[1];
    const float* bp = (const float*)d_in[2];
    const float* w  = (const float*)d_in[3];
    const float* u  = (const float*)d_in[4];
    const float* b  = (const float*)d_in[5];
    const float* Wr = (const float*)d_in[6];
    const float* br = (const float*)d_in[7];
    float* out = (float*)d_out;

    const int T = GT;
    int Bn = in_sizes[0] / T;                          // 4096
    // ws layout: [0, 2048) padded Wr (16*KPAD floats), [2048, ...) padded feat
    size_t wrpBytes  = 2048;
    size_t featBytes = (size_t)Bn * KPAD * sizeof(float);
    int useWmma = (ws_size >= wrpBytes + featBytes) ? 1 : 0;
    float* Wrp  = (float*)d_ws;
    float* feat = (float*)((char*)d_ws + wrpBytes);

    if (useWmma)
        genesis_pad_wr<<<1, 320, 0, stream>>>(Wr, Wrp);

    int threads = Bn * 2;                              // (batch, hemi)
    int blk = 64;
    genesis_scan<<<(threads + blk - 1) / blk, blk, 0, stream>>>(
        x, Wp, bp, w, u, b, Wr, br, feat, out, Bn, useWmma ? 0 : 1);

    if (useWmma) {
        int waves  = (Bn + 15) / 16;                   // 256
        int blocks = (waves * 32 + 255) / 256;         // 32
        genesis_readout_wmma<<<blocks, 256, 0, stream>>>(feat, Wrp, br, out, Bn);
    }
}
